// FourierLayer_56925496541710
// MI455X (gfx1250) — matrix operand order repaired and checked by reading, not compiled, not run on hardware
//
#include <hip/hip_runtime.h>
#include <hip/hip_bf16.h>

typedef __attribute__((ext_vector_type(16))) _Float16 v16h;
typedef __attribute__((ext_vector_type(8)))  float    v8f;

#define T_DIM 2048
#define D_DIM 512
#define B_DIM 32
#define NBIN  1023   // bins m = 1..1023
#define TOPK  16

union Frag { v16h v; uint4 u[2]; };

// ---------------------------------------------------------------------------
// Kernel 0a: build DFT basis, f16, row-major [2048][2048].
// rows 0..1023   :  cos(2*pi*(r+1)*t/2048)      -> Re part of rfft
// rows 1024..2047: -sin(2*pi*(r-1023)*t/2048)   -> Im part of rfft
// Angle reduced exactly via (m*t) mod 2048 (power-of-two period).
// ---------------------------------------------------------------------------
__global__ void fl_basis(_Float16* __restrict__ basisH) {
  int g = blockIdx.x * 256 + threadIdx.x;        // 0 .. 2048*2048-1
  int r = g >> 11;
  int t = g & 2047;
  int m = (r < 1024) ? (r + 1) : (r - 1023);
  int ph = (m * t) & 2047;
  float ang = (float)ph * (6.28318530717958647692f / 2048.0f);
  float s, c;
  __sincosf(ang, &s, &c);
  basisH[g] = (_Float16)((r < 1024) ? c : -s);
}

// ---------------------------------------------------------------------------
// Kernel 0b: transpose + convert x[b][t][d] f32 -> xTh[b][d][t] f16
// so GEMM B-fragments are K-contiguous loads.
// ---------------------------------------------------------------------------
__global__ void fl_xpose(const float* __restrict__ x, _Float16* __restrict__ xTh) {
  __shared__ float tile[32][33];
  const int b = blockIdx.z;
  const int tBase = blockIdx.x * 32;
  const int dBase = blockIdx.y * 32;
  const int tx = threadIdx.x, ty = threadIdx.y;
#pragma unroll
  for (int r = 0; r < 4; ++r) {
    int tt = ty + 8 * r;
    tile[tt][tx] = x[((size_t)b * T_DIM + tBase + tt) * D_DIM + dBase + tx];
  }
  __syncthreads();
#pragma unroll
  for (int r = 0; r < 4; ++r) {
    int dd = ty + 8 * r;
    xTh[((size_t)b * D_DIM + dBase + dd) * T_DIM + tBase + tx] = (_Float16)tile[tx][dd];
  }
}

// ---------------------------------------------------------------------------
// Kernel 1: DFT-as-GEMM with v_wmma_f32_16x16x32_f16.
//   spec[b][d][r] = sum_t basis[r][t] * x[b][t][d]   (r=0..2047, f32 accum)
// Workgroup: 256 threads = 8 waves, wave tile 64x32 (M x N), WG tile 128x128.
// Grid: (2048/128, 512/128, 32) = (16, 4, 32).
// A/B per-lane layouts per CDNA5 ISA 7.12.2 (16-bit, 16x16x32).
// ---------------------------------------------------------------------------
__global__ __launch_bounds__(256) void fl_gemm(const _Float16* __restrict__ basisH,
                                               const _Float16* __restrict__ xTh,
                                               float* __restrict__ spec) {
  const int lane = threadIdx.x & 31;
  const int w    = threadIdx.x >> 5;      // 0..7
  const int wm   = w & 1;                 // 2 M-waves
  const int wn   = w >> 1;                // 4 N-waves
  const int b    = blockIdx.z;
  const int M0   = blockIdx.x * 128 + wm * 64;
  const int N0   = blockIdx.y * 128 + wn * 32;
  const int r0   = lane & 15;
  const int kbA  = (lane < 16) ? 0 : 8;   // A: lanes<16 K{0-7,16-23}, else K{8-15,24-31}
  const int kbB  = (lane < 16) ? 0 : 8;   // B: lanes<16 K{0-7,16-23}, else K{8-15,24-31}

  const _Float16* aRow[4];
#pragma unroll
  for (int i = 0; i < 4; ++i)
    aRow[i] = basisH + (size_t)(M0 + 16 * i + r0) * T_DIM + kbA;
  const _Float16* bRow[2];
#pragma unroll
  for (int j = 0; j < 2; ++j)
    bRow[j] = xTh + ((size_t)b * D_DIM + (N0 + 16 * j + r0)) * T_DIM + kbB;

  v8f acc[4][2] = {};

  for (int k0 = 0; k0 < T_DIM; k0 += 32) {
    Frag a[4], bf[2];
#pragma unroll
    for (int i = 0; i < 4; ++i) {
      a[i].u[0] = *(const uint4*)(aRow[i] + k0);        // K = kbA .. kbA+7
      a[i].u[1] = *(const uint4*)(aRow[i] + k0 + 16);   // K = kbA+16 .. kbA+23
    }
#pragma unroll
    for (int j = 0; j < 2; ++j) {
      bf[j].u[0] = *(const uint4*)(bRow[j] + k0);       // K = kbB .. kbB+7
      bf[j].u[1] = *(const uint4*)(bRow[j] + k0 + 16);  // K = kbB+16 .. kbB+23
    }
#pragma unroll
    for (int i = 0; i < 4; ++i)
#pragma unroll
      for (int j = 0; j < 2; ++j)
        acc[i][j] = __builtin_amdgcn_wmma_f32_16x16x32_f16(
            false, a[i].v, false, bf[j].v, (short)0, acc[i][j], false, false);
  }

  // Store: C element e -> row M0+16i+e(+8 for lanes>=16), col N0+16j+(lane&15).
  const int hi8 = (lane < 16) ? 0 : 8;
#pragma unroll
  for (int i = 0; i < 4; ++i) {
#pragma unroll
    for (int j = 0; j < 2; ++j) {
      const int d = N0 + 16 * j + r0;
      float* o = spec + ((size_t)b * D_DIM + d) * T_DIM + (M0 + 16 * i + hi8);
#pragma unroll
      for (int e = 0; e < 8; ++e) o[e] = acc[i][j][e];
    }
  }
}

// ---------------------------------------------------------------------------
// Kernel 2: per-(b,d) top-16 bins by |spec|^2; one wave32 per block.
// spec row layout: [0..1022] = Re(m=1..1023), [1024..2046] = Im(m=1..1023).
// Emits coefficients re, im, m (amp*cos(ph)=re, amp*sin(ph)=im -> no atan2).
// ---------------------------------------------------------------------------
__global__ void fl_topk(const float* __restrict__ spec,
                        float* __restrict__ cr, float* __restrict__ ci,
                        float* __restrict__ cm) {
  __shared__ float mags[1024];
  __shared__ float sv[32];
  __shared__ int   si[32];
  __shared__ int   sel[TOPK];
  const int lane = threadIdx.x;
  const int d = blockIdx.x, b = blockIdx.y;
  const float* row = spec + ((size_t)b * D_DIM + d) * T_DIM;

#pragma unroll 4
  for (int i = 0; i < 32; ++i) {
    int idx = lane + 32 * i;              // idx = m-1
    float m2 = -1.0f;
    if (idx < NBIN) {
      float re = row[idx];
      float im = row[1024 + idx];
      m2 = re * re + im * im;
    }
    mags[idx] = m2;
  }
  for (int rr = 0; rr < TOPK; ++rr) {
    __syncthreads();
    float best = -2.0f; int bi = 0;
    for (int i = 0; i < 32; ++i) {
      int idx = lane + 32 * i;
      float v = mags[idx];
      if (v > best) { best = v; bi = idx; }
    }
    sv[lane] = best; si[lane] = bi;
    __syncthreads();
    if (lane == 0) {
      float bb = -3.0f; int bidx = 0;
      for (int l = 0; l < 32; ++l)
        if (sv[l] > bb) { bb = sv[l]; bidx = si[l]; }
      sel[rr] = bidx;
      mags[bidx] = -4.0f;                 // knock out for next round
    }
  }
  __syncthreads();
  if (lane < TOPK) {
    int idx = sel[lane];
    size_t o = ((size_t)b * D_DIM + d) * TOPK + lane;
    cr[o] = row[idx];
    ci[o] = row[1024 + idx];
    cm[o] = (float)(idx + 1);             // m
  }
}

// ---------------------------------------------------------------------------
// Kernel 3: synthesis via Chebyshev recurrence g_t = 2cos(w) g_{t-1} - g_{t-2}
// with g seeded per 256-step t-chunk. 1 FMA per (term, t). Stores coalesced
// over d (lane-contiguous).  y[b][t][d] = sum_k re*cos(w t) - im*sin(w t),
// w = 2*pi*m/(2048*2047)  (linspace(0,1,T) has step 1/2047).
// Grid: (2, 8, 32) x 256 threads -> d = bx*256+tid, t0 = by*256, b = bz.
// ---------------------------------------------------------------------------
__global__ __launch_bounds__(256) void fl_synth(const float* __restrict__ cr,
                                                const float* __restrict__ ci,
                                                const float* __restrict__ cm,
                                                float* __restrict__ out) {
  const int d  = blockIdx.x * 256 + threadIdx.x;
  const int t0 = blockIdx.y * 256;
  const int b  = blockIdx.z;
  const size_t bd = (size_t)b * D_DIM + d;
  const float W = 6.28318530717958647692f / (2048.0f * 2047.0f);

  float p[TOPK], q[TOPK], c2[TOPK];
#pragma unroll
  for (int k = 0; k < TOPK; ++k) {
    float m  = cm[bd * TOPK + k];
    float re = cr[bd * TOPK + k];
    float im = ci[bd * TOPK + k];
    float wk = m * W;                       // <= ~pi/2048 per step
    float sw, cw; __sincosf(wk, &sw, &cw);
    float a0 = wk * (float)t0;              // <= ~pi total
    float s0, c0; __sincosf(a0, &s0, &c0);
    p[k]  = re * c0 - im * s0;              // value at t0
    float c1 = c0 * cw - s0 * sw;
    float s1 = s0 * cw + c0 * sw;
    q[k]  = re * c1 - im * s1;              // value at t0+1
    c2[k] = 2.0f * cw;
  }

  float* o = out + ((size_t)b * T_DIM + t0) * D_DIM + d;
  for (int tt = 0; tt < 256; tt += 2) {
    float sum0 = 0.0f, sum1 = 0.0f;
#pragma unroll
    for (int k = 0; k < TOPK; ++k) {
      sum0 += p[k];
      sum1 += q[k];
      float n0 = __builtin_fmaf(c2[k], q[k], -p[k]);   // t+2
      float n1 = __builtin_fmaf(c2[k], n0,  -q[k]);    // t+3
      p[k] = n0; q[k] = n1;
    }
    o[(size_t)tt * D_DIM]       = sum0;
    o[(size_t)(tt + 1) * D_DIM] = sum1;
  }
}

// ---------------------------------------------------------------------------
// Launch
// ---------------------------------------------------------------------------
extern "C" void kernel_launch(void* const* d_in, const int* in_sizes, int n_in,
                              void* d_out, int out_size, void* d_ws, size_t ws_size,
                              hipStream_t stream) {
  (void)in_sizes; (void)n_in; (void)out_size; (void)ws_size;
  const float* x = (const float*)d_in[0];
  float* out = (float*)d_out;

  // Workspace layout (bytes):
  //   spec   : 32*512*2048*4        = 134217728
  //   coefRe : 16384*16*4           =   1048576
  //   coefIm : 16384*16*4           =   1048576
  //   coefM  : 16384*16*4           =   1048576
  //   basisH : 2048*2048*2          =   8388608
  //   xTh    : 32*512*2048*2        =  67108864   (total ~203 MiB)
  char* w = (char*)d_ws;
  float*    spec   = (float*)w;
  float*    coefRe = (float*)(w + 134217728ull);
  float*    coefIm = coefRe + (size_t)B_DIM * D_DIM * TOPK;
  float*    coefM  = coefIm + (size_t)B_DIM * D_DIM * TOPK;
  _Float16* basisH = (_Float16*)(w + 134217728ull + 3ull * 1048576ull);
  _Float16* xTh    = basisH + (size_t)T_DIM * T_DIM;

  fl_basis<<<(T_DIM * T_DIM) / 256, 256, 0, stream>>>(basisH);
  fl_xpose<<<dim3(T_DIM / 32, D_DIM / 32, B_DIM), dim3(32, 8), 0, stream>>>(x, xTh);
  fl_gemm <<<dim3(T_DIM / 128, D_DIM / 128, B_DIM), 256, 0, stream>>>(basisH, xTh, spec);
  fl_topk <<<dim3(D_DIM, B_DIM), 32, 0, stream>>>(spec, coefRe, coefIm, coefM);
  fl_synth<<<dim3(D_DIM / 256, T_DIM / 256, B_DIM), 256, 0, stream>>>(coefRe, coefIm, coefM, out);
}
